// GRU_14018773254529
// MI455X (gfx1250) — matrix operand order repaired and checked by reading, not compiled, not run on hardware
//
#include <hip/hip_runtime.h>

// ---------------------------------------------------------------------------
// GRU (B=4096, T=512, H=64, I=1) for gfx1250 (MI455X).
// Recurrent matmuls via v_wmma_f32_16x16x32_bf16 (f32 accumulate).
// Each wave owns a 16-row batch tile; W_hh bf16 B-fragments become
// register-resident (compiler hoists the loop-invariant LDS loads);
// h round-trips through LDS (bf16) each step for C-layout -> A-layout.
// Gate nonlinearities use native v_tanh_f32 when available (1 TRANS op
// per sigmoid/tanh instead of exp+rcp = 2 TRANS ops).
// ---------------------------------------------------------------------------

typedef __attribute__((ext_vector_type(16))) __bf16 v16bf;
typedef __attribute__((ext_vector_type(8)))  float  v8f;
typedef __attribute__((ext_vector_type(4)))  float  f32x4;

#define HID   64
#define TLEN  512
#define TC    32          // x chunk (timesteps) staged in LDS
#define WAVES 4           // waves per block
#define ROWS  16          // batch rows per wave (WMMA M)

// LDS geometry
#define FRAG_LANE_STRIDE 48           // bytes per lane per B fragment (16B-aligned, 2-way max)
#define FRAG_BYTES (32 * FRAG_LANE_STRIDE)
#define NFRAG 24                      // 3 gates * 4 n-tiles * 2 k-chunks
#define HROW_STRIDE 144               // bytes per h row (64 bf16 = 128B used, padded)

#if __has_builtin(__builtin_amdgcn_tanhf)
__device__ __forceinline__ float fast_tanh(float x) {
    return __builtin_amdgcn_tanhf(x);                  // v_tanh_f32
}
__device__ __forceinline__ float fast_sigmoid(float x) {
    // sigmoid(x) = 0.5*tanh(x/2) + 0.5  -> 1 TRANS + 2 VALU
    return __builtin_amdgcn_tanhf(0.5f * x) * 0.5f + 0.5f;
}
#else
__device__ __forceinline__ float fast_sigmoid(float x) {
    float e = __builtin_amdgcn_exp2f(-1.4426950408889634f * x);
    return __builtin_amdgcn_rcpf(1.0f + e);
}
__device__ __forceinline__ float fast_tanh(float x) {
    float e = __builtin_amdgcn_exp2f(2.8853900817779268f * x);
    return 1.0f - 2.0f * __builtin_amdgcn_rcpf(e + 1.0f);
}
#endif

__device__ __forceinline__ v8f wmma_bf16(v16bf a, v16bf b, v8f c) {
    return __builtin_amdgcn_wmma_f32_16x16x32_bf16(
        /*neg_a=*/false, a, /*neg_b=*/false, b,
        /*c_mod=*/(short)0, c, /*reuse_a=*/false, /*reuse_b=*/false);
}

// B fragment: per-lane 8 dwords at p, p+16
__device__ __forceinline__ v16bf load_bfrag(const unsigned char* p) {
    union { f32x4 q[2]; v16bf v; } u;
    u.q[0] = *(const f32x4*)(p);
    u.q[1] = *(const f32x4*)(p + 16);
    return u.v;
}
// A fragment: per-lane 8 dwords at p (K=kbase..+7) and p+32 (K=kbase+16..+23)
__device__ __forceinline__ v16bf load_afrag(const unsigned char* p) {
    union { f32x4 q[2]; v16bf v; } u;
    u.q[0] = *(const f32x4*)(p);
    u.q[1] = *(const f32x4*)(p + 32);
    return u.v;
}

__global__ __launch_bounds__(WAVES * 32, 1)
void gru_wmma_kernel(const float* __restrict__ x,
                     const float* __restrict__ W_ih,
                     const float* __restrict__ W_hh,
                     const float* __restrict__ b_ih,
                     const float* __restrict__ b_hh,
                     const float* __restrict__ W_out,
                     const float* __restrict__ b_out,
                     float* __restrict__ out)
{
    __shared__ __align__(16) unsigned char s_wfrag[NFRAG * FRAG_BYTES];        // 36864 B
    __shared__ __align__(16) float         s_x[WAVES][TC][16];                 //  8192 B
    __shared__ __align__(16) unsigned char s_h[WAVES][ROWS * HROW_STRIDE];     //  9216 B
    __shared__ __align__(16) float         s_out[WAVES][16][16];               //  4096 B

    const int tid  = threadIdx.x;
    const int wave = tid >> 5;
    const int lane = tid & 31;
    const int nl   = lane & 15;        // column-within-tile / row index
    const int hl   = lane >> 4;        // lane half
    const int b0   = blockIdx.x * (WAVES * ROWS) + wave * ROWS;

    // ---- one-time: build W_hh bf16 B-fragments in LDS (shared by block) ----
    // result[m,n] = sum_k h[m,k] * W_hh[g*64+n, k]  => B[k,n] = W_hh[g*64+n, k]
    // 32x16 bf16 B layout: lane half hl holds K=8hl..8hl+7 then K=16+8hl..16+8hl+7.
    for (int f = wave; f < NFRAG; f += WAVES) {
        const int g  = f >> 3;           // f = g*8 + nt*2 + kc
        const int nt = (f >> 1) & 3;
        const int kc = f & 1;
        const int n  = nt * 16 + nl;
        const int k0 = kc * 32 + hl * 8;
        const float* src = W_hh + (g * 64 + n) * 64 + k0;
        union { f32x4 q[2]; __bf16 e[16]; } u;
        #pragma unroll
        for (int i = 0; i < 16; ++i) u.e[i] = (__bf16)src[i < 8 ? i : i + 8];
        unsigned char* dst = s_wfrag + f * FRAG_BYTES + lane * FRAG_LANE_STRIDE;
        *(f32x4*)(dst)      = u.q[0];
        *(f32x4*)(dst + 16) = u.q[1];
    }

    // ---- per-lane input-projection coefficients (I=1) and biases ----
    float wih_r[4], wih_z[4], wih_n[4];
    float bias_r[4], bias_z[4], bih_n[4], bhh_n[4], wo[4];
    #pragma unroll
    for (int nt = 0; nt < 4; ++nt) {
        const int n = nt * 16 + nl;
        wih_r[nt]  = W_ih[n];
        bias_r[nt] = b_ih[n] + b_hh[n];
        wih_z[nt]  = W_ih[64 + n];
        bias_z[nt] = b_ih[64 + n] + b_hh[64 + n];
        wih_n[nt]  = W_ih[128 + n];
        bih_n[nt]  = b_ih[128 + n];
        bhh_n[nt]  = b_hh[128 + n];
        wo[nt]     = W_out[n];
    }

    // ---- init h = 0 (registers, C layout) and its LDS bf16 mirror ----
    v8f h[4];
    #pragma unroll
    for (int nt = 0; nt < 4; ++nt)
        #pragma unroll
        for (int j = 0; j < 8; ++j) h[nt][j] = 0.0f;
    {
        unsigned char* hrow = s_h[wave] + nl * HROW_STRIDE + hl * 64;
        f32x4 z4 = {0.0f, 0.0f, 0.0f, 0.0f};
        *(f32x4*)(hrow)      = z4;
        *(f32x4*)(hrow + 16) = z4;
        *(f32x4*)(hrow + 32) = z4;
        *(f32x4*)(hrow + 48) = z4;
    }
    __syncthreads();

    // ---- recurrence over T ----
    for (int t = 0; t < TLEN; ++t) {
        if ((t & (TC - 1)) == 0) {
            // stage x[:, t:t+TC] for this wave's rows, transposed -> s_x[t][row]
            const int row = lane >> 1;
            #pragma unroll
            for (int i = 0; i < 4; ++i) {
                const int toff = (lane & 1) * 4 + i * 8;
                f32x4 v = *(const f32x4*)(x + (size_t)(b0 + row) * TLEN + t + toff);
                s_x[wave][toff + 0][row] = v.x;
                s_x[wave][toff + 1][row] = v.y;
                s_x[wave][toff + 2][row] = v.z;
                s_x[wave][toff + 3][row] = v.w;
            }
        }
        const int tc = t & (TC - 1);

        // x value for row m = j + 8*hl  (broadcast reads, conflict-free)
        f32x4 xa = *(const f32x4*)&s_x[wave][tc][hl * 8];
        f32x4 xb = *(const f32x4*)&s_x[wave][tc][hl * 8 + 4];
        float xrow[8] = {xa.x, xa.y, xa.z, xa.w, xb.x, xb.y, xb.z, xb.w};

        // A fragments of h (bf16, row-major in LDS). 16-bit A 16x32 layout:
        // lane half hl -> K += 8; regs 4..7 -> K += 16.
        const unsigned char* hbase = s_h[wave] + nl * HROW_STRIDE;
        v16bf a0 = load_afrag(hbase + hl * 16);        // K-chunk 0 (K 0..31)
        v16bf a1 = load_afrag(hbase + 64 + hl * 16);   // K-chunk 1 (K 32..63)

        // ---- r gate ----
        v8f r[4];
        #pragma unroll
        for (int nt = 0; nt < 4; ++nt) {
            v8f c;
            #pragma unroll
            for (int j = 0; j < 8; ++j) c[j] = xrow[j] * wih_r[nt] + bias_r[nt];
            const unsigned char* bf = s_wfrag + (0 * 8 + nt * 2) * FRAG_BYTES + lane * FRAG_LANE_STRIDE;
            c = wmma_bf16(a0, load_bfrag(bf), c);
            c = wmma_bf16(a1, load_bfrag(bf + FRAG_BYTES), c);
            #pragma unroll
            for (int j = 0; j < 8; ++j) r[nt][j] = fast_sigmoid(c[j]);
        }

        // ---- n gate: n = tanh(xn + r * (h@Whn^T + bhn)) ----
        v8f nv[4];
        #pragma unroll
        for (int nt = 0; nt < 4; ++nt) {
            v8f c;
            #pragma unroll
            for (int j = 0; j < 8; ++j) c[j] = bhh_n[nt];
            const unsigned char* bf = s_wfrag + (2 * 8 + nt * 2) * FRAG_BYTES + lane * FRAG_LANE_STRIDE;
            c = wmma_bf16(a0, load_bfrag(bf), c);
            c = wmma_bf16(a1, load_bfrag(bf + FRAG_BYTES), c);
            #pragma unroll
            for (int j = 0; j < 8; ++j) {
                const float xn = xrow[j] * wih_n[nt] + bih_n[nt];
                nv[nt][j] = fast_tanh(xn + r[nt][j] * c[j]);
            }
        }

        // ---- z gate + state update: h = n + z*(h - n) ----
        #pragma unroll
        for (int nt = 0; nt < 4; ++nt) {
            v8f c;
            #pragma unroll
            for (int j = 0; j < 8; ++j) c[j] = xrow[j] * wih_z[nt] + bias_z[nt];
            const unsigned char* bf = s_wfrag + (1 * 8 + nt * 2) * FRAG_BYTES + lane * FRAG_LANE_STRIDE;
            c = wmma_bf16(a0, load_bfrag(bf), c);
            c = wmma_bf16(a1, load_bfrag(bf + FRAG_BYTES), c);
            #pragma unroll
            for (int j = 0; j < 8; ++j) {
                const float z = fast_sigmoid(c[j]);
                h[nt][j] = nv[nt][j] + z * (h[nt][j] - nv[nt][j]);
            }
        }

        // ---- write h back to LDS as bf16 (C layout -> row-major [m][k]) ----
        // C tile element: lane half hl, VGPR j -> m = j + 8*hl, n = nt*16 + nl
        #pragma unroll
        for (int nt = 0; nt < 4; ++nt)
            #pragma unroll
            for (int j = 0; j < 8; ++j) {
                __bf16* p = (__bf16*)(s_h[wave] + (j + 8 * hl) * HROW_STRIDE) + nt * 16 + nl;
                *p = (__bf16)h[nt][j];
            }
    }

    // ---- out[b] = h[b,:] . W_out + b_out ----
    #pragma unroll
    for (int j = 0; j < 8; ++j) {
        float p = h[0][j] * wo[0] + h[1][j] * wo[1] + h[2][j] * wo[2] + h[3][j] * wo[3];
        s_out[wave][j + 8 * hl][nl] = p;   // partial over 4 columns per lane
    }
    __syncthreads();
    if (lane < 16) {
        const f32x4* rp = (const f32x4*)s_out[wave][lane];
        f32x4 s4 = rp[0];
        #pragma unroll
        for (int q = 1; q < 4; ++q) {
            f32x4 v = rp[q];
            s4.x += v.x; s4.y += v.y; s4.z += v.z; s4.w += v.w;
        }
        out[b0 + lane] = (s4.x + s4.y) + (s4.z + s4.w) + b_out[0];
    }
}

extern "C" void kernel_launch(void* const* d_in, const int* in_sizes, int n_in,
                              void* d_out, int out_size, void* d_ws, size_t ws_size,
                              hipStream_t stream) {
    (void)n_in; (void)d_ws; (void)ws_size; (void)in_sizes;
    const float* x     = (const float*)d_in[0];
    const float* W_ih  = (const float*)d_in[1];
    const float* W_hh  = (const float*)d_in[2];
    const float* b_ih  = (const float*)d_in[3];
    const float* b_hh  = (const float*)d_in[4];
    const float* W_out = (const float*)d_in[5];
    const float* b_out = (const float*)d_in[6];
    float* out = (float*)d_out;

    const int B = out_size;                 // 4096
    const int rowsPerBlock = WAVES * ROWS;  // 64
    const int blocks = (B + rowsPerBlock - 1) / rowsPerBlock;  // 64
    gru_wmma_kernel<<<blocks, WAVES * 32, 0, stream>>>(
        x, W_ih, W_hh, b_ih, b_hh, W_out, b_out, out);
}
